// KoreanSLLM_88734024335734
// MI455X (gfx1250) — matrix operand order repaired and checked by reading, not compiled, not run on hardware
//
#include <hip/hip_runtime.h>
#include <math.h>

// Problem constants (match reference).
#define BB   4
#define SS   1024
#define DD   1024
#define HH   16
#define HDD  64
#define DFFC 4096
#define LL   12
#define VV   32000

// GEMM block tiling.
#define MBLK 64
#define NBLK 256
#define KBLK 32

typedef __bf16 bf16;
typedef __attribute__((ext_vector_type(16))) __bf16 v16bf;
typedef __attribute__((ext_vector_type(8)))  __bf16 v8bf;
typedef __attribute__((ext_vector_type(8)))  float  v8f;

// ---------------------------------------------------------------------------
// WMMA helper (CDNA5: D = A(16x32 bf16) * B(32x16 bf16) + C(16x16 f32))
// ---------------------------------------------------------------------------
__device__ __forceinline__ v8f wmma_bf16(v16bf a, v16bf b, v8f c) {
  // 8 args: (neg_a, A, neg_b, B, c_mod, C, reuse_a, reuse_b)
  return __builtin_amdgcn_wmma_f32_16x16x32_bf16(false, a, false, b, (short)0, c,
                                                 false, false);
}

// A-matrix 16x32 bf16 fragment (ISA 7.12.2):
//   lanes 0-15 : row M = lane,    K = k0+{0..7} (v0-3) and k0+{16..23} (v4-7)
//   lanes 16-31: row M = lane-16, K offsets shifted +8
__device__ __forceinline__ v16bf load_a_frag(const bf16* __restrict__ A, int lda,
                                             int m0, int k0, int lane) {
  int m  = m0 + (lane & 15);
  int kb = k0 + ((lane & 16) >> 1);            // +0 or +8
  const bf16* p = A + (size_t)m * lda + kb;
  v8bf lo = *(const v8bf*)p;                   // K = kb .. kb+7
  v8bf hi = *(const v8bf*)(p + 16);            // K = kb+16 .. kb+23
  v16bf r;
#pragma unroll
  for (int i = 0; i < 8; ++i) { r[i] = lo[i]; r[8 + i] = hi[i]; }
  return r;
}

// B-matrix 32x16 bf16 fragment. Bt stored [N][K] row-major, so a B-column
// (fixed n) is contiguous along K. lanes 0-15: K=k0+{0..7},{16..23}; lanes 16-31: +8.
__device__ __forceinline__ v16bf load_b_frag(const bf16* __restrict__ Bt, int ldbt,
                                             int n0, int k0, int lane) {
  int n  = n0 + (lane & 15);
  int kb = k0 + ((lane & 16) >> 1);            // +0 or +8
  const bf16* p = Bt + (size_t)n * ldbt + kb;
  v8bf lo = *(const v8bf*)p;
  v8bf hi = *(const v8bf*)(p + 16);
  v16bf r;
#pragma unroll
  for (int i = 0; i < 8; ++i) { r[i] = lo[i]; r[8 + i] = hi[i]; }
  return r;
}

// ---------------------------------------------------------------------------
// CDNA5 async global->LDS copy (ASYNCcnt-tracked, CDNA5 ISA).
// LDS aperture keeps the byte offset in addr[31:0], so the low 32 bits of a
// generic shared pointer are the VDST LDS address.
// ---------------------------------------------------------------------------
__device__ __forceinline__ unsigned lds_addr32(const void* p) {
  return (unsigned)(uintptr_t)p;
}
__device__ __forceinline__ void async_copy_b128(void* lds_dst, const bf16* gsrc) {
  unsigned l = lds_addr32(lds_dst);
  asm volatile("global_load_async_to_lds_b128 %0, %1, off"
               :: "v"(l), "v"(gsrc) : "memory");
}
__device__ __forceinline__ void wait_asynccnt0() {
  asm volatile("s_wait_asynccnt 0x0" ::: "memory");
}

// ---------------------------------------------------------------------------
// Weight fp32 -> bf16 converters (transpose variant stores [N][K]).
// ---------------------------------------------------------------------------
__global__ __launch_bounds__(256) void conv_t_kernel(const float* __restrict__ src,
                                                     bf16* __restrict__ dst,
                                                     int K, int N) {
  size_t total = (size_t)K * N;
  for (size_t i = (size_t)blockIdx.x * 256 + threadIdx.x; i < total;
       i += (size_t)gridDim.x * 256) {
    size_t k = i / (size_t)N, n = i % (size_t)N;
    dst[n * (size_t)K + k] = (bf16)src[i];
  }
}

__global__ __launch_bounds__(256) void conv_kernel(const float* __restrict__ src,
                                                   bf16* __restrict__ dst,
                                                   size_t total) {
  for (size_t i = (size_t)blockIdx.x * 256 + threadIdx.x; i < total;
       i += (size_t)gridDim.x * 256)
    dst[i] = (bf16)src[i];
}

// ---------------------------------------------------------------------------
// Embedding + sinusoidal positional encoding -> fp32 residual stream.
// ---------------------------------------------------------------------------
__global__ __launch_bounds__(256) void embed_kernel(const int* __restrict__ ids,
                                                    const float* __restrict__ emb,
                                                    float* __restrict__ x) {
  int row = blockIdx.x;                 // b*S + pos
  int pos = row & (SS - 1);
  int id  = ids[row];
  const float* e = emb + (size_t)id * DD;
  for (int d = threadIdx.x; d < DD; d += 256) {
    int i2 = d & ~1;
    float freq = __expf(-(float)i2 * (9.210340372f / (float)DD));  // ln(10000)
    float ang  = (float)pos * freq;
    float pe   = (d & 1) ? __cosf(ang) : __sinf(ang);
    x[(size_t)row * DD + d] = e[d] + pe;
  }
}

// ---------------------------------------------------------------------------
// LayerNorm (fp32 in, bf16 out). One block per row of D=1024.
// ---------------------------------------------------------------------------
__global__ __launch_bounds__(256) void ln_kernel(const float* __restrict__ x,
                                                 const float* __restrict__ gamma,
                                                 const float* __restrict__ beta,
                                                 bf16* __restrict__ out) {
  int row = blockIdx.x;
  const float* xr = x + (size_t)row * DD;
  float s = 0.f, ss = 0.f;
  for (int i = threadIdx.x; i < DD; i += 256) {
    float v = xr[i]; s += v; ss += v * v;
  }
#pragma unroll
  for (int o = 16; o > 0; o >>= 1) {
    s  += __shfl_down(s,  o, 32);
    ss += __shfl_down(ss, o, 32);
  }
  __shared__ float rs[8], rss[8];
  __shared__ float mu_s, rstd_s;
  if ((threadIdx.x & 31) == 0) { rs[threadIdx.x >> 5] = s; rss[threadIdx.x >> 5] = ss; }
  __syncthreads();
  if (threadIdx.x == 0) {
    float S = 0.f, Q = 0.f;
    for (int w = 0; w < 8; ++w) { S += rs[w]; Q += rss[w]; }
    float mu  = S / (float)DD;
    float var = Q / (float)DD - mu * mu;
    mu_s = mu; rstd_s = rsqrtf(var + 1e-5f);
  }
  __syncthreads();
  float mu = mu_s, rstd = rstd_s;
  for (int i = threadIdx.x; i < DD; i += 256)
    out[(size_t)row * DD + i] = (bf16)((xr[i] - mu) * rstd * gamma[i] + beta[i]);
}

// ---------------------------------------------------------------------------
// bf16 WMMA GEMM with async-LDS double buffering.
//   C[M,N] = A[M,K] @ Bt[N,K]^T  (+bias, +residual, +GELU)
// Block = 8 waves (2 Mx4 N), block tile 64x256, K stepped by 32.
// Per K-step: stage next 64x32 A-slab and 256x32 B-slab into LDS with
// global_load_async_to_lds_b128 while WMMAs consume the current slab.
// TRANS_V: store bf16 output per-head transposed, (b*S+s, h*64+hd) ->
//          Vt[((b*H + h)*64 + hd)*S + s]  (feeds contiguous P@V fragments).
// ---------------------------------------------------------------------------
template <bool OUT_BF16, bool BIAS, bool RES, bool GELU, bool TRANS_V>
__global__ __launch_bounds__(256) void gemm_wmma(const bf16* __restrict__ A,
                                                 const bf16* __restrict__ Bt,
                                                 const float* __restrict__ bias,
                                                 const float* __restrict__ res,
                                                 void* __restrict__ Cout,
                                                 int M, int N, int K) {
  __shared__ bf16 sA[2][MBLK][KBLK];   //  2 x 4 KB
  __shared__ bf16 sB[2][NBLK][KBLK];   //  2 x 16 KB
  int tid  = threadIdx.x;
  int lane = tid & 31;
  int wave = tid >> 5;
  int wm = wave >> 2;                  // 0..1  (32-row slice)
  int wn = wave & 3;                   // 0..3  (64-col slice)
  int m0 = blockIdx.y * MBLK;
  int n0 = blockIdx.x * NBLK;

  auto stage = [&](int buf, int k0) {
    // A slab: 64x32 bf16 = 256 chunks of 16B, one per thread.
    {
      int row = tid >> 2, c = tid & 3;
      async_copy_b128(&sA[buf][row][c * 8],
                      A + (size_t)(m0 + row) * K + k0 + c * 8);
    }
    // B slab: 256x32 bf16 = 1024 chunks of 16B, four per thread.
#pragma unroll
    for (int i = 0; i < 4; ++i) {
      int id = i * 256 + tid;
      int row = id >> 2, c = id & 3;
      async_copy_b128(&sB[buf][row][c * 8],
                      Bt + (size_t)(n0 + row) * K + k0 + c * 8);
    }
  };

  v8f acc[2][4] = {};
  int nk = K / KBLK;
  stage(0, 0);
  for (int kk = 0; kk < nk; ++kk) {
    int cur = kk & 1;
    wait_asynccnt0();                  // our slab copies have landed
    __syncthreads();                   // everyone's copies have landed
    if (kk + 1 < nk) stage(cur ^ 1, (kk + 1) * KBLK);
    v16bf af0 = load_a_frag(&sA[cur][0][0], KBLK, wm * 32 + 0,  0, lane);
    v16bf af1 = load_a_frag(&sA[cur][0][0], KBLK, wm * 32 + 16, 0, lane);
#pragma unroll
    for (int t = 0; t < 4; ++t) {
      v16bf bfr = load_b_frag(&sB[cur][0][0], KBLK, wn * 64 + t * 16, 0, lane);
      acc[0][t] = wmma_bf16(af0, bfr, acc[0][t]);
      acc[1][t] = wmma_bf16(af1, bfr, acc[1][t]);
    }
    __syncthreads();                   // compute done before slab reuse
  }

  // C layout: VGPR r -> row +r (+8 for lanes 16-31), col = lane&15.
  int colb = lane & 15;
  int rloc = (lane & 16) >> 1;
#pragma unroll
  for (int sm = 0; sm < 2; ++sm) {
    int rowb = m0 + wm * 32 + sm * 16 + rloc;
#pragma unroll
    for (int t = 0; t < 4; ++t) {
      int col = n0 + wn * 64 + t * 16 + colb;
      float bv = BIAS ? bias[col] : 0.f;
#pragma unroll
      for (int r = 0; r < 8; ++r) {
        int row = rowb + r;
        float v = acc[sm][t][r] + bv;
        if (GELU) v = 0.5f * v * (1.f + erff(v * 0.70710678f));
        if (RES)  v += res[(size_t)row * N + col];
        if (TRANS_V) {
          // row = b*S + s, col = h*64 + hd  ->  Vt[((b*H + h)*64 + hd)*S + s]
          int bb_ = row >> 10, s_ = row & (SS - 1);
          int h_ = col >> 6, hd_ = col & (HDD - 1);
          ((bf16*)Cout)[(((size_t)bb_ * HH + h_) * HDD + hd_) * SS + s_] = (bf16)v;
        } else if (OUT_BF16) {
          ((bf16*)Cout)[(size_t)row * N + col] = (bf16)v;
        } else {
          ((float*)Cout)[(size_t)row * N + col] = v;
        }
      }
    }
  }
}

// ---------------------------------------------------------------------------
// Attention: one wave per (b, h, 16-query tile). Scores via WMMA (causal tile
// skip), full-row softmax in LDS, then probs(bf16) @ Vt via WMMA. Vt is the
// per-head transposed V ([b][h][hd][s]), so P@V B-columns are contiguous.
// ---------------------------------------------------------------------------
__global__ __launch_bounds__(32) void attn_kernel(const bf16* __restrict__ q,
                                                  const bf16* __restrict__ k,
                                                  const bf16* __restrict__ vt,
                                                  const int* __restrict__ ids,
                                                  bf16* __restrict__ o) {
  __shared__ float sc[16 * SS];                       // 64 KB score buffer
  int lane = threadIdx.x;
  int m0 = blockIdx.x * 16;
  int h  = blockIdx.y;
  int b  = blockIdx.z;
  const bf16* qb = q  + (size_t)b * SS * DD + h * HDD;
  const bf16* kb = k  + (size_t)b * SS * DD + h * HDD;
  const bf16* vb = vt + ((size_t)b * HH + h) * HDD * SS;   // [hd][s]

  // Q fragments for the two K-steps of HD=64.
  v16bf aq0 = load_a_frag(qb, DD, m0, 0, lane);
  v16bf aq1 = load_a_frag(qb, DD, m0, 32, lane);

  int colb = lane & 15;
  int rloc = (lane & 16) >> 1;                        // 0 or 8
  int jmax = m0 >> 4;                                 // causal tile limit
  for (int j = 0; j <= jmax; ++j) {
    v8f s = {};
    s = wmma_bf16(aq0, load_b_frag(kb, DD, j * 16, 0, lane), s);
    s = wmma_bf16(aq1, load_b_frag(kb, DD, j * 16, 32, lane), s);
    int key = j * 16 + colb;
    bool padok = ids[b * SS + key] != 0;
#pragma unroll
    for (int r = 0; r < 8; ++r) {
      int qrow = m0 + rloc + r;
      float val = (padok && key <= qrow) ? s[r] * 0.125f : -__builtin_inff();
      sc[(rloc + r) * SS + key] = val;
    }
  }
  __syncthreads();

  // Softmax over keys [0, limit). Two lanes cooperate per row.
  int row   = lane & 15;
  int halfs = lane >> 4;
  int limit = m0 + 16;
  float mx = -__builtin_inff();
  for (int c = halfs; c < limit; c += 2) mx = fmaxf(mx, sc[row * SS + c]);
  mx = fmaxf(mx, __shfl_xor(mx, 16, 32));
  float sum = 0.f;
  for (int c = halfs; c < limit; c += 2) {
    float e = __expf(sc[row * SS + c] - mx);
    sc[row * SS + c] = e;
    sum += e;
  }
  sum += __shfl_xor(sum, 16, 32);
  float inv = 1.f / sum;
  for (int c = halfs; c < limit; c += 2) sc[row * SS + c] *= inv;
  // Zero-pad to a 32-key boundary for the PV WMMA loop.
  int limit32 = (limit + 31) & ~31;
  if (limit32 != limit) {
    for (int i = lane; i < 16 * 16; i += 32)
      sc[(i >> 4) * SS + limit + (i & 15)] = 0.f;
  }
  __syncthreads();

  // O = P @ V : K-dim = keys, N = 64 head dims (4 tiles of 16).
  // B-columns (fixed hd over keys) are contiguous rows of Vt.
  v8f ao[4] = {};
  int kkend = limit32 >> 5;
  for (int kk = 0; kk < kkend; ++kk) {
    int arow = lane & 15;
    int kbof = kk * 32 + ((lane & 16) >> 1);
    const float* sp = &sc[arow * SS + kbof];
    v16bf a;
#pragma unroll
    for (int i = 0; i < 8; ++i) { a[i] = (bf16)sp[i]; a[8 + i] = (bf16)sp[16 + i]; }
#pragma unroll
    for (int t = 0; t < 4; ++t) {
      v16bf bbf = load_b_frag(vb, SS, t * 16, kk * 32, lane);
      ao[t] = wmma_bf16(a, bbf, ao[t]);
    }
  }
  int rowb = m0 + rloc;
#pragma unroll
  for (int t = 0; t < 4; ++t)
#pragma unroll
    for (int r = 0; r < 8; ++r)
      o[((size_t)b * SS + rowb + r) * DD + h * HDD + t * 16 + colb] = (bf16)ao[t][r];
}

// ---------------------------------------------------------------------------
// Host orchestration.
// ---------------------------------------------------------------------------
extern "C" void kernel_launch(void* const* d_in, const int* in_sizes, int n_in,
                              void* d_out, int out_size, void* d_ws, size_t ws_size,
                              hipStream_t stream) {
  (void)in_sizes; (void)n_in; (void)out_size; (void)ws_size;
  const int*   ids   = (const int*)  d_in[0];
  const float* emb   = (const float*)d_in[1];
  const float* wq    = (const float*)d_in[2];
  const float* wk    = (const float*)d_in[3];
  const float* wv    = (const float*)d_in[4];
  const float* wo_w  = (const float*)d_in[5];
  const float* wo_b  = (const float*)d_in[6];
  const float* ln1_s = (const float*)d_in[7];
  const float* ln1_b = (const float*)d_in[8];
  const float* ln2_s = (const float*)d_in[9];
  const float* ln2_b = (const float*)d_in[10];
  const float* ff1_w = (const float*)d_in[11];
  const float* ff1_b = (const float*)d_in[12];
  const float* ff2_w = (const float*)d_in[13];
  const float* ff2_b = (const float*)d_in[14];
  const float* lnf_s = (const float*)d_in[15];
  const float* lnf_b = (const float*)d_in[16];

  const int M = BB * SS;                      // 4096 rows
  char* ws = (char*)d_ws;
  size_t off = 0;
  bf16*  WBUF = (bf16*)(ws + off); off += (size_t)VV * DD * 2;      // 64 MB reused weight slab
  float* X    = (float*)(ws + off); off += (size_t)M * DD * 4;      // fp32 residual stream
  bf16*  Hb   = (bf16*)(ws + off); off += (size_t)M * DD * 2;       // LN output (bf16)
  bf16*  Qb   = (bf16*)(ws + off); off += (size_t)M * DD * 2;
  bf16*  Kb   = (bf16*)(ws + off); off += (size_t)M * DD * 2;
  bf16*  Vt   = (bf16*)(ws + off); off += (size_t)M * DD * 2;       // per-head transposed V
  bf16*  Ob   = (bf16*)(ws + off); off += (size_t)M * DD * 2;
  bf16*  FFH  = (bf16*)(ws + off); off += (size_t)M * DFFC * 2;

  dim3 gD(DD / NBLK,   M / MBLK);             // (4, 64)
  dim3 gF(DFFC / NBLK, M / MBLK);             // (16, 64)
  dim3 gV(VV / NBLK,   M / MBLK);             // (125, 64)
  dim3 gA(SS / 16, HH, BB);

  embed_kernel<<<M, 256, 0, stream>>>(ids, emb, X);

  for (int l = 0; l < LL; ++l) {
    // --- attention sub-block ---
    ln_kernel<<<M, 256, 0, stream>>>(X, ln1_s + (size_t)l * DD, ln1_b + (size_t)l * DD, Hb);
    conv_t_kernel<<<4096, 256, 0, stream>>>(wq + (size_t)l * DD * DD, WBUF, DD, DD);
    gemm_wmma<true, false, false, false, false><<<gD, 256, 0, stream>>>(Hb, WBUF, nullptr, nullptr, Qb, M, DD, DD);
    conv_t_kernel<<<4096, 256, 0, stream>>>(wk + (size_t)l * DD * DD, WBUF, DD, DD);
    gemm_wmma<true, false, false, false, false><<<gD, 256, 0, stream>>>(Hb, WBUF, nullptr, nullptr, Kb, M, DD, DD);
    conv_t_kernel<<<4096, 256, 0, stream>>>(wv + (size_t)l * DD * DD, WBUF, DD, DD);
    gemm_wmma<true, false, false, false, true><<<gD, 256, 0, stream>>>(Hb, WBUF, nullptr, nullptr, Vt, M, DD, DD);
    attn_kernel<<<gA, 32, 0, stream>>>(Qb, Kb, Vt, ids, Ob);
    conv_t_kernel<<<4096, 256, 0, stream>>>(wo_w + (size_t)l * DD * DD, WBUF, DD, DD);
    gemm_wmma<false, true, true, false, false><<<gD, 256, 0, stream>>>(Ob, WBUF, wo_b + (size_t)l * DD, X, X, M, DD, DD);
    // --- FFN sub-block ---
    ln_kernel<<<M, 256, 0, stream>>>(X, ln2_s + (size_t)l * DD, ln2_b + (size_t)l * DD, Hb);
    conv_t_kernel<<<4096, 256, 0, stream>>>(ff1_w + (size_t)l * DD * DFFC, WBUF, DD, DFFC);
    gemm_wmma<true, true, false, true, false><<<gF, 256, 0, stream>>>(Hb, WBUF, ff1_b + (size_t)l * DFFC, nullptr, FFH, M, DFFC, DD);
    conv_t_kernel<<<4096, 256, 0, stream>>>(ff2_w + (size_t)l * DFFC * DD, WBUF, DFFC, DD);
    gemm_wmma<false, true, true, false, false><<<gD, 256, 0, stream>>>(FFH, WBUF, ff2_b + (size_t)l * DD, X, X, M, DD, DFFC);
  }

  // Final LN + tied lm-head (Bt = emb [V][D], plain convert).
  ln_kernel<<<M, 256, 0, stream>>>(X, lnf_s, lnf_b, Hb);
  conv_kernel<<<4096, 256, 0, stream>>>(emb, WBUF, (size_t)VV * DD);
  gemm_wmma<false, false, false, false, false><<<gV, 256, 0, stream>>>(Hb, WBUF, nullptr, nullptr, d_out, M, VV, DD);
}
